// DynamicEdgeBiasAttention_71605694759111
// MI455X (gfx1250) — matrix operand order repaired and checked by reading, not compiled, not run on hardware
//
#include <hip/hip_runtime.h>
#include <hip/hip_bf16.h>

// Problem constants (fixed by the reference harness)
#define BDIM 2
#define NDIM 2048
#define DDIM 256
#define HDIM 8
#define DH   32

typedef __attribute__((ext_vector_type(16))) __bf16 v16bf;
typedef __attribute__((ext_vector_type(8)))  float  v8f;

// ---------------------------------------------------------------------------
// WMMA fragment loaders; A and B share one contiguous k order (k = half*16 + s).
// A-matrix 16x32 bf16: lane row = lane%16; VGPR i packs K = half*16 + 2i, 2i+1;
//                      (half=lane/16)
// B-matrix 32x16 bf16: lane col = lane%16; VGPR i packs K-rows half*16+2i, +1.
// C/D 16x16 f32:       lane col = lane%16; VGPR i holds row i + 8*half.
// ---------------------------------------------------------------------------

__device__ __forceinline__ v16bf load_a_frag_bf16(const __bf16* __restrict__ base, int lda) {
  const int lane  = threadIdx.x & 31;
  const int lhalf = lane >> 4;
  const int row   = lane & 15;
  v16bf a;
#pragma unroll
  for (int i = 0; i < 8; ++i) {
    const int k = lhalf * 16 + 2 * i;
    a[2 * i]     = base[row * lda + k];
    a[2 * i + 1] = base[row * lda + k + 1];
  }
  return a;
}

// A-fragment sourced from an f32 LDS tile (16 x lda), converting to bf16.
__device__ __forceinline__ v16bf load_a_frag_f32(const float* base, int lda) {
  const int lane  = threadIdx.x & 31;
  const int lhalf = lane >> 4;
  const int row   = lane & 15;
  v16bf a;
#pragma unroll
  for (int i = 0; i < 8; ++i) {
    const int k = lhalf * 16 + 2 * i;
    a[2 * i]     = (__bf16)base[row * lda + k];
    a[2 * i + 1] = (__bf16)base[row * lda + k + 1];
  }
  return a;
}

// B[k][col] = base[col*ld + k]  (i.e. the stored matrix transposed; K-pairs contiguous)
__device__ __forceinline__ v16bf load_bT_frag_bf16(const __bf16* base, int ld) {
  const int lane  = threadIdx.x & 31;
  const int lhalf = lane >> 4;
  const int col   = lane & 15;
  v16bf b;
#pragma unroll
  for (int i = 0; i < 8; ++i) {
    const int k = lhalf * 16 + 2 * i;
    b[2 * i]     = base[col * ld + k];
    b[2 * i + 1] = base[col * ld + k + 1];
  }
  return b;
}

// B[k][col] = base[k*ld + col]  (row-major stored matrix used directly as B)
__device__ __forceinline__ v16bf load_b_frag_bf16(const __bf16* base, int ld) {
  const int lane  = threadIdx.x & 31;
  const int lhalf = lane >> 4;
  const int col   = lane & 15;
  v16bf b;
#pragma unroll
  for (int i = 0; i < 8; ++i) {
    const int k = lhalf * 16 + 2 * i;
    b[2 * i]     = base[k * ld + col];
    b[2 * i + 1] = base[(k + 1) * ld + col];
  }
  return b;
}

__device__ __forceinline__ v8f wmma_bf16(v16bf a, v16bf b, v8f c) {
  return __builtin_amdgcn_wmma_f32_16x16x32_bf16(false, a, false, b, (short)0, c,
                                                 false, false);
}

// 16-lane (half-wave) reductions; xor masks < 16 never cross the half boundary.
__device__ __forceinline__ float rmax16(float v) {
#pragma unroll
  for (int off = 1; off < 16; off <<= 1) v = fmaxf(v, __shfl_xor(v, off, 32));
  return v;
}
__device__ __forceinline__ float rsum16(float v) {
#pragma unroll
  for (int off = 1; off < 16; off <<= 1) v += __shfl_xor(v, off, 32);
  return v;
}

// Per-edge bias MLP: b2h + sum_k w2[k] * leaky_relu(a*w1[k] + b1[k])
__device__ __forceinline__ float edge_bias(float a, const float* w1, const float* bb1,
                                           const float* w2, float b2h) {
  float r = b2h;
#pragma unroll
  for (int k = 0; k < 8; ++k) {
    float t = fmaf(a, w1[k], bb1[k]);
    t = (t > 0.f) ? t : 0.01f * t;
    r = fmaf(w2[k], t, r);
  }
  return r;
}

// ---------------------------------------------------------------------------
// CDNA5 async global->LDS lane copy (16B per lane), tracked with ASYNCcnt.
// CDNA5 ISA opcode 98 / CDNA5 ISA.
// ---------------------------------------------------------------------------
__device__ __forceinline__ void async_copy_b128(void* lds_dst, const void* gsrc) {
  const unsigned laddr = (unsigned)(size_t)lds_dst;            // low 32b = LDS offset
  const unsigned long long gaddr = (unsigned long long)(size_t)gsrc;
  asm volatile("global_load_async_to_lds_b128 %0, %1, off"
               :: "v"(laddr), "v"(gaddr) : "memory");
}
#define WAIT_ASYNCCNT_0() asm volatile("s_wait_asynccnt 0x0" ::: "memory")
#define WAIT_ASYNCCNT_2() asm volatile("s_wait_asynccnt 0x2" ::: "memory")

// ---------------------------------------------------------------------------
// Kernel 0: f32 -> bf16 cast
// ---------------------------------------------------------------------------
__global__ void cast_f32_bf16(const float* __restrict__ src, __bf16* __restrict__ dst,
                              int n) {
  int i = blockIdx.x * blockDim.x + threadIdx.x;
  if (i < n) dst[i] = (__bf16)src[i];
}

// ---------------------------------------------------------------------------
// Kernel 1: QKV projections.  One wave computes one 16x16 tile of x @ W^T + b,
// scattering the result into the (B,H,N,32) bf16 head layout.
// grid = (B*N/16, D/16, 3)
// ---------------------------------------------------------------------------
__global__ void qkv_proj_wmma(const __bf16* __restrict__ xb,
                              const __bf16* __restrict__ Wq,
                              const __bf16* __restrict__ Wk,
                              const __bf16* __restrict__ Wv,
                              const float* __restrict__ bq,
                              const float* __restrict__ bk,
                              const float* __restrict__ bv,
                              __bf16* __restrict__ Qo, __bf16* __restrict__ Ko,
                              __bf16* __restrict__ Vo) {
  const __bf16* W;
  const float* bias;
  __bf16* out;
  if (blockIdx.z == 0)      { W = Wq; bias = bq; out = Qo; }
  else if (blockIdx.z == 1) { W = Wk; bias = bk; out = Ko; }
  else                      { W = Wv; bias = bv; out = Vo; }

  const int rowbase = blockIdx.x * 16;  // token tile over B*N
  const int colbase = blockIdx.y * 16;  // output-feature tile over D

  v8f acc = {};
#pragma unroll
  for (int kk = 0; kk < DDIM; kk += 32) {
    v16bf a = load_a_frag_bf16(xb + (size_t)rowbase * DDIM + kk, DDIM);
    v16bf b = load_bT_frag_bf16(W + (size_t)colbase * DDIM + kk, DDIM);
    acc = wmma_bf16(a, b, acc);
  }

  const int lane  = threadIdx.x & 31;
  const int lhalf = lane >> 4;
  const int col   = lane & 15;
  const int cg    = colbase + col;
  const int h     = cg >> 5;   // head
  const int d     = cg & 31;   // dim within head
  const float bb  = bias[cg];
#pragma unroll
  for (int i = 0; i < 8; ++i) {
    const int grow = rowbase + i + 8 * lhalf;     // flat token index in [0, B*N)
    const int b0   = grow >> 11;                  // / NDIM
    const int n    = grow & (NDIM - 1);           // % NDIM
    out[(((size_t)(b0 * HDIM + h) * NDIM + n) * DH) + d] = (__bf16)(acc[i] + bb);
  }
}

// ---------------------------------------------------------------------------
// Kernel 2: fused flash attention with dynamic edge bias.
// Block = 128 threads (4 waves) covering 64 query rows of one (b,h).
// 32-key K/V chunks are staged in LDS via async global->LDS copies (ASYNCcnt),
// double-buffered so DMA of chunk c+1 overlaps WMMA+softmax of chunk c.
// Never materializes scores or the (B,H,N,N) bias tensor.
// grid = (N/64, H, B), block = 128
// ---------------------------------------------------------------------------
__global__ void flash_attn_bias(const __bf16* __restrict__ Q,
                                const __bf16* __restrict__ K,
                                const __bf16* __restrict__ V,
                                const float* __restrict__ A,
                                const float* __restrict__ W1,
                                const float* __restrict__ b1,
                                const float* __restrict__ W2,
                                const float* __restrict__ b2,
                                __bf16* __restrict__ O) {
  __shared__ __align__(16) __bf16 ktile[2][32 * DH];   // 2 x 2KB
  __shared__ __align__(16) __bf16 vtile[2][32 * DH];   // 2 x 2KB
  __shared__ float pstage[4][16 * 32];                 // per-wave P staging, 8KB

  const int b     = blockIdx.z;
  const int h     = blockIdx.y;
  const int tid   = threadIdx.x;        // 0..127
  const int wave  = tid >> 5;           // 0..3
  const int lane  = tid & 31;
  const int lhalf = lane >> 4;
  const int col   = lane & 15;
  const int qbase = blockIdx.x * 64 + wave * 16;

  const __bf16* Qp = Q + ((size_t)(b * HDIM + h) * NDIM + qbase) * DH;
  const __bf16* Kh = K + (size_t)(b * HDIM + h) * NDIM * DH;
  const __bf16* Vh = V + (size_t)(b * HDIM + h) * NDIM * DH;
  const float*  Ap = A + ((size_t)b * NDIM + qbase) * NDIM;
  float* ps = pstage[wave];

  const v16bf qf = load_a_frag_bf16(Qp, DH);

  float w1[8], bb1[8], w2[8];
#pragma unroll
  for (int k = 0; k < 8; ++k) {
    w1[k]  = W1[k];
    bb1[k] = b1[k];
    w2[k]  = W2[h * HDIM + k];
  }
  const float b2h   = b2[h];
  const float scale = 0.17677669529663687f;  // 1/sqrt(32)

  v8f acc0 = {}, acc1 = {};
  float m[8], l[8];
#pragma unroll
  for (int i = 0; i < 8; ++i) { m[i] = -3.0e38f; l[i] = 0.f; }

  const int nchunks = NDIM / 32;  // 64

  // Prologue: async-stage chunk 0 into buffer 0 (one b128 lane-copy each for K,V)
  async_copy_b128(&ktile[0][tid * 8], Kh + (size_t)tid * 8);
  async_copy_b128(&vtile[0][tid * 8], Vh + (size_t)tid * 8);

  for (int c = 0; c < nchunks; ++c) {
    const int buf   = c & 1;
    const int jbase = c * 32;

    if (c + 1 < nchunks) {
      // Kick off next chunk's DMA into the other buffer, then wait for ours.
      const size_t nxt = (size_t)(jbase + 32) * DH;
      async_copy_b128(&ktile[buf ^ 1][tid * 8], Kh + nxt + (size_t)tid * 8);
      async_copy_b128(&vtile[buf ^ 1][tid * 8], Vh + nxt + (size_t)tid * 8);
      WAIT_ASYNCCNT_2();   // async loads complete in order: chunk c is resident
    } else {
      WAIT_ASYNCCNT_0();
    }
    __syncthreads();       // every wave's slice of chunk c is in LDS

    // keep the adjacency rows warm (global_prefetch_b8)
    __builtin_prefetch(Ap + jbase + 64, 0, 1);

    // S = Q K^T for two 16-key subtiles (K dim = d_h = 32, one WMMA each)
    const __bf16* kt = &ktile[buf][0];
    v16bf k0 = load_bT_frag_bf16(kt, DH);
    v16bf k1 = load_bT_frag_bf16(kt + 16 * DH, DH);
    v8f s0 = {}, s1 = {};
    s0 = wmma_bf16(qf, k0, s0);
    s1 = wmma_bf16(qf, k1, s1);

#pragma unroll
    for (int i = 0; i < 8; ++i) {
      const int row = i + 8 * lhalf;
      const float* ar = Ap + (size_t)row * NDIM + jbase + col;
      const float a0 = ar[0];
      const float a1 = ar[16];
      float s0v = fmaf(s0[i], scale, edge_bias(a0, w1, bb1, w2, b2h));
      float s1v = fmaf(s1[i], scale, edge_bias(a1, w1, bb1, w2, b2h));

      // online softmax update for this row
      float mx = rmax16(fmaxf(s0v, s1v));
      float mn = fmaxf(m[i], mx);
      float alpha = __expf(m[i] - mn);
      float e0 = __expf(s0v - mn);
      float e1 = __expf(s1v - mn);
      float sum = rsum16(e0 + e1);
      l[i] = l[i] * alpha + sum;
      m[i] = mn;
      acc0[i] *= alpha;
      acc1[i] *= alpha;

      // per-wave staging: same-wave LDS ops are DScnt-ordered, no barrier needed
      ps[row * 32 + col]      = e0;
      ps[row * 32 + 16 + col] = e1;
    }

    // O += P V   (A = P 16x32 from LDS, B = V 32x16 for each 16-col half)
    v16bf pa = load_a_frag_f32(ps, 32);
    const __bf16* vt = &vtile[buf][0];
    v16bf v0 = load_b_frag_bf16(vt, DH);
    v16bf v1 = load_b_frag_bf16(vt + 16, DH);
    acc0 = wmma_bf16(pa, v0, acc0);
    acc1 = wmma_bf16(pa, v1, acc1);

    __syncthreads();  // all waves done reading buf before it is DMA'd again
  }

  // normalize and scatter into (B,N,D) bf16 layout for the output projection
#pragma unroll
  for (int i = 0; i < 8; ++i) {
    const int row = i + 8 * lhalf;
    const float inv = 1.0f / l[i];
    const size_t o = ((size_t)b * NDIM + qbase + row) * DDIM + h * DH;
    O[o + col]      = (__bf16)(acc0[i] * inv);
    O[o + 16 + col] = (__bf16)(acc1[i] * inv);
  }
}

// ---------------------------------------------------------------------------
// Kernel 3: output projection, f32 result.  grid = (B*N/16, D/16)
// ---------------------------------------------------------------------------
__global__ void out_proj_wmma(const __bf16* __restrict__ Ob,
                              const __bf16* __restrict__ Wo,
                              const float* __restrict__ bo,
                              float* __restrict__ out) {
  const int rowbase = blockIdx.x * 16;
  const int colbase = blockIdx.y * 16;

  v8f acc = {};
#pragma unroll
  for (int kk = 0; kk < DDIM; kk += 32) {
    v16bf a = load_a_frag_bf16(Ob + (size_t)rowbase * DDIM + kk, DDIM);
    v16bf b = load_bT_frag_bf16(Wo + (size_t)colbase * DDIM + kk, DDIM);
    acc = wmma_bf16(a, b, acc);
  }

  const int lane  = threadIdx.x & 31;
  const int lhalf = lane >> 4;
  const int col   = lane & 15;
  const float bb  = bo[colbase + col];
#pragma unroll
  for (int i = 0; i < 8; ++i) {
    const int grow = rowbase + i + 8 * lhalf;
    out[(size_t)grow * DDIM + colbase + col] = acc[i] + bb;
  }
}

// ---------------------------------------------------------------------------
// Host-side launcher
// ---------------------------------------------------------------------------
extern "C" void kernel_launch(void* const* d_in, const int* in_sizes, int n_in,
                              void* d_out, int out_size, void* d_ws, size_t ws_size,
                              hipStream_t stream) {
  (void)in_sizes; (void)n_in; (void)out_size; (void)ws_size;
  const float* x  = (const float*)d_in[0];
  const float* A  = (const float*)d_in[1];
  const float* Wq = (const float*)d_in[2];
  const float* bq = (const float*)d_in[3];
  const float* Wk = (const float*)d_in[4];
  const float* bk = (const float*)d_in[5];
  const float* Wv = (const float*)d_in[6];
  const float* bv = (const float*)d_in[7];
  const float* Wo = (const float*)d_in[8];
  const float* bo = (const float*)d_in[9];
  const float* W1 = (const float*)d_in[10];
  const float* b1 = (const float*)d_in[11];
  const float* W2 = (const float*)d_in[12];
  const float* b2 = (const float*)d_in[13];
  float* out = (float*)d_out;

  const size_t nx = (size_t)BDIM * NDIM * DDIM;  // 1,048,576
  const size_t nw = (size_t)DDIM * DDIM;         // 65,536

  char* ws = (char*)d_ws;
  __bf16* xb  = (__bf16*)ws; ws += nx * 2;
  __bf16* Wqb = (__bf16*)ws; ws += nw * 2;
  __bf16* Wkb = (__bf16*)ws; ws += nw * 2;
  __bf16* Wvb = (__bf16*)ws; ws += nw * 2;
  __bf16* Wob = (__bf16*)ws; ws += nw * 2;
  __bf16* Qb  = (__bf16*)ws; ws += nx * 2;
  __bf16* Kb  = (__bf16*)ws; ws += nx * 2;
  __bf16* Vb  = (__bf16*)ws; ws += nx * 2;
  __bf16* Ob  = (__bf16*)ws; ws += nx * 2;

  cast_f32_bf16<<<(int)((nx + 255) / 256), 256, 0, stream>>>(x, xb, (int)nx);
  cast_f32_bf16<<<(int)((nw + 255) / 256), 256, 0, stream>>>(Wq, Wqb, (int)nw);
  cast_f32_bf16<<<(int)((nw + 255) / 256), 256, 0, stream>>>(Wk, Wkb, (int)nw);
  cast_f32_bf16<<<(int)((nw + 255) / 256), 256, 0, stream>>>(Wv, Wvb, (int)nw);
  cast_f32_bf16<<<(int)((nw + 255) / 256), 256, 0, stream>>>(Wo, Wob, (int)nw);

  dim3 gq((BDIM * NDIM) / 16, DDIM / 16, 3);
  qkv_proj_wmma<<<gq, 32, 0, stream>>>(xb, Wqb, Wkb, Wvb, bq, bk, bv, Qb, Kb, Vb);

  dim3 gf(NDIM / 64, HDIM, BDIM);
  flash_attn_bias<<<gf, 128, 0, stream>>>(Qb, Kb, Vb, A, W1, b1, W2, b2, Ob);

  dim3 go((BDIM * NDIM) / 16, DDIM / 16);
  out_proj_wmma<<<go, 32, 0, stream>>>(Ob, Wob, bo, out);
}
